// NeuralSDE_18786186953185
// MI455X (gfx1250) — matrix operand order repaired and checked by reading, not compiled, not run on hardware
//
#include <hip/hip_runtime.h>
#include <stdint.h>

// ---------------------------------------------------------------------------
// Types for CDNA5 WMMA (wave32, 16x16x32 bf16 -> f32)
// ---------------------------------------------------------------------------
typedef __bf16 bf16_t;
typedef __attribute__((ext_vector_type(16))) __bf16 v16bf;
typedef __attribute__((ext_vector_type(8)))  __bf16 bf16x8;
typedef __attribute__((ext_vector_type(8)))  float  v8f;

#define BATCH 4096
#define TSTEPS 64
#define HDIM 128
#define NDIM 32
#define WDIM 256
#define DOUT 16

static __device__ __forceinline__ bf16_t f2bf(float x) {
  union { float f; uint32_t u; } v; v.f = x;
  uint32_t r = v.u + 0x7FFFu + ((v.u >> 16) & 1u);   // round-to-nearest-even
  union { unsigned short s; bf16_t b; } o; o.s = (unsigned short)(r >> 16);
  return o.b;
}
static __device__ __forceinline__ float bf2f(bf16_t x) {
  union { unsigned short s; bf16_t b; } u; u.b = x;
  union { uint32_t i; float f; } o; o.i = ((uint32_t)u.s) << 16;
  return o.f;
}

// ---------------------------------------------------------------------------
// bf16 WMMA GEMM:  C[M,N] = act( A[M,K] @ W[N,K]^T + bias + t*tcol ) * scale
// Templated tiling:
//   block = BM x 128, 256 threads = 8 waves arranged WR x WC,
//   wave tile = (FM*16) x (FN*16)  ->  FM*FN WMMAs per 32-K chunk.
//   K chunks of 32, double-buffered LDS; next chunk's global->VGPR loads are
//   issued before the WMMAs so they are in flight during compute.
// Instantiations:
//   <64, 2,4, 2,2>  : 64x128 block, wave 32x32 (4 wmma)  - small-N layers
//   <128,4,2, 2,4>  : 128x128 block, wave 32x64 (8 wmma) - N=4096 layer (82% FLOPs)
// ---------------------------------------------------------------------------
#define GBK 32
#define LDT 40              // padded LDS row stride (bf16), 80B: 16B-aligned b128

template<int BM, int WR, int WC, int FM, int FN>
__global__ __launch_bounds__(256) void gemm_bf16_kernel(
    const bf16_t* __restrict__ A, int lda,
    const bf16_t* __restrict__ W, int ldw,
    const float* __restrict__ bias,
    const float* __restrict__ tcol,         // optional: weight column for time input
    const float* __restrict__ ts, int tidx, // t value = ts[tidx] (device-side)
    const float* __restrict__ scale,        // optional per-output-column scale
    int M, int N, int K, int act,           // act: 0=none 1=relu 2=tanh
    float* __restrict__ outF, bf16_t* __restrict__ outB, int ldo)
{
  constexpr int BN  = 128;
  constexpr int ASZ = BM * LDT;
  constexpr int BSZ = BN * LDT;
  constexpr int AIT = (BM * GBK) / (256 * 8);  // bf16x8 pieces per thread (A)
  constexpr int BIT = (BN * GBK) / (256 * 8);  // bf16x8 pieces per thread (B)

  __shared__ bf16_t As[2 * ASZ];
  __shared__ bf16_t Bs[2 * BSZ];

  const int bm   = blockIdx.y * BM;
  const int bn   = blockIdx.x * BN;
  const int tid  = threadIdx.x;
  const int lane = tid & 31;
  const int wave = tid >> 5;
  const int wm   = (wave / WC) * (FM * 16);
  const int wn   = (wave % WC) * (FN * 16);
  const int laneLo = lane & 15;
  const int laneHi = lane >> 4;

  const float tval = (tcol != nullptr) ? ts[tidx] : 0.0f;
  const int nk = K / GBK;

  v8f acc[FM][FN] = {};

  // ---- prologue: stage chunk 0 straight into LDS buffer 0 ----
#pragma unroll
  for (int i = 0; i < AIT; i++) {
    int e = (tid + i * 256) * 8, r = e >> 5, cc = e & 31;
    *(bf16x8*)(As + r * LDT + cc) =
        *(const bf16x8*)(A + (size_t)(bm + r) * lda + cc);
  }
#pragma unroll
  for (int i = 0; i < BIT; i++) {
    int e = (tid + i * 256) * 8, r = e >> 5, cc = e & 31;
    *(bf16x8*)(Bs + r * LDT + cc) =
        *(const bf16x8*)(W + (size_t)(bn + r) * ldw + cc);
  }
  __syncthreads();

  for (int kc = 0; kc < nk; kc++) {
    const int buf = kc & 1;
    const bool more = (kc + 1) < nk;

    // issue next chunk's global loads (in flight during the WMMAs below)
    bf16x8 ra[AIT], rb[BIT];
    if (more) {
      const int k1 = (kc + 1) * GBK;
#pragma unroll
      for (int i = 0; i < AIT; i++) {
        int e = (tid + i * 256) * 8, r = e >> 5, cc = e & 31;
        ra[i] = *(const bf16x8*)(A + (size_t)(bm + r) * lda + k1 + cc);
      }
#pragma unroll
      for (int i = 0; i < BIT; i++) {
        int e = (tid + i * 256) * 8, r = e >> 5, cc = e & 31;
        rb[i] = *(const bf16x8*)(W + (size_t)(bn + r) * ldw + k1 + cc);
      }
      if (kc + 2 < nk)   // prefetch chunk after next (global_prefetch_b8)
        __builtin_prefetch(W + (size_t)(bn + (tid >> 1)) * ldw +
                           (kc + 2) * GBK + (tid & 1) * 16, 0, 1);
    }

    // ---- fragments from LDS buffer `buf` ----
    // A frag (16-bit A 16x32): row=lane%16; elems 0..7 -> K=laneHi*8+j,
    //                                       elems 8..15 -> K=16+laneHi*8+j
    // B frag (16-bit B 32x16): col=lane%16; elems 0..7 -> K=laneHi*8+j,
    //                                       elems 8..15 -> K=16+laneHi*8+j
    const bf16_t* aBase = As + buf * ASZ;
    const bf16_t* bBase = Bs + buf * BSZ;
    v16bf fa[FM], fb[FN];
#pragma unroll
    for (int f = 0; f < FM; f++) {
      const bf16x8* ap = (const bf16x8*)(aBase + (wm + f * 16 + laneLo) * LDT);
      bf16x8 lo = ap[laneHi], hi = ap[2 + laneHi];
#pragma unroll
      for (int i = 0; i < 8; i++) { fa[f][i] = lo[i]; fa[f][8 + i] = hi[i]; }
    }
#pragma unroll
    for (int f = 0; f < FN; f++) {
      const bf16x8* bp = (const bf16x8*)(bBase + (wn + f * 16 + laneLo) * LDT);
      bf16x8 lo = bp[laneHi], hi = bp[2 + laneHi];
#pragma unroll
      for (int i = 0; i < 8; i++) { fb[f][i] = lo[i]; fb[f][8 + i] = hi[i]; }
    }

#pragma unroll
    for (int fm = 0; fm < FM; fm++)
#pragma unroll
      for (int fn = 0; fn < FN; fn++)
        acc[fm][fn] = __builtin_amdgcn_wmma_f32_16x16x32_bf16(
            false, fa[fm], false, fb[fn], (short)0, acc[fm][fn], false, false);

    // spill next chunk into the alternate LDS buffer
    if (more) {
      bf16_t* aDst = As + (buf ^ 1) * ASZ;
      bf16_t* bDst = Bs + (buf ^ 1) * BSZ;
#pragma unroll
      for (int i = 0; i < AIT; i++) {
        int e = (tid + i * 256) * 8, r = e >> 5, cc = e & 31;
        *(bf16x8*)(aDst + r * LDT + cc) = ra[i];
      }
#pragma unroll
      for (int i = 0; i < BIT; i++) {
        int e = (tid + i * 256) * 8, r = e >> 5, cc = e & 31;
        *(bf16x8*)(bDst + r * LDT + cc) = rb[i];
      }
    }
    __syncthreads();
  }

  // ---- epilogue ----
  // C layout: VGPR r -> row = laneHi*8 + r, col = lane%16 (per 16x16 tile)
#pragma unroll
  for (int fn = 0; fn < FN; fn++) {
    const int col = bn + wn + fn * 16 + laneLo;
    float b = bias ? bias[col] : 0.0f;
    if (tcol) b += tval * tcol[col];
    const float s = scale ? scale[col] : 1.0f;
#pragma unroll
    for (int fm = 0; fm < FM; fm++) {
      const int rowBase = bm + wm + fm * 16 + laneHi * 8;
#pragma unroll
      for (int r = 0; r < 8; r++) {
        float v = acc[fm][fn][r] + b;
        if (act == 1)      v = fmaxf(v, 0.0f);
        else if (act == 2) v = tanhf(v);
        v *= s;
        size_t o = (size_t)(rowBase + r) * ldo + col;
        if (outF) outF[o] = v;
        if (outB) outB[o] = f2bf(v);
      }
    }
  }
}

// ---------------------------------------------------------------------------
// Elementwise / small kernels
// ---------------------------------------------------------------------------
__global__ void cvt_kernel(bf16_t* dst, const float* src, int n) {
  int i = blockIdx.x * blockDim.x + threadIdx.x;
  if (i < n) dst[i] = f2bf(src[i]);
}

// split W[N][129] -> col0[N] (f32, time column) + wy[N][128] (bf16)
__global__ void split_kernel(bf16_t* wy, float* col0, const float* w, int N) {
  int i = blockIdx.x * blockDim.x + threadIdx.x;
  if (i >= N * 128) return;
  int n = i >> 7, k = i & 127;
  wy[i] = f2bf(w[n * 129 + 1 + k]);
  if (k == 0) col0[n] = w[n * 129];
}

__global__ void copyf_kernel(float* dst, const float* src, int n) {
  int i = blockIdx.x * blockDim.x + threadIdx.x;
  if (i < n) dst[i] = src[i];
}

// predictor: yhat1 = 2y - yhat + d0*dt + g0.dw ; emits f32 + bf16 + gdw0
__global__ void pre_kernel(const bf16_t* __restrict__ g0,
                           const float* __restrict__ dW, int step,
                           const float* __restrict__ ts,
                           const float* __restrict__ y,
                           const float* __restrict__ yh,
                           const float* __restrict__ d0,
                           float* __restrict__ yh_out,
                           bf16_t* __restrict__ yb_out,
                           float* __restrict__ gdw_out) {
  int idx = blockIdx.x * blockDim.x + threadIdx.x;   // b*128 + h
  if (idx >= BATCH * HDIM) return;
  int b = idx >> 7, h = idx & 127;
  const bf16_t* g  = g0 + (size_t)b * (HDIM * NDIM) + h * NDIM;
  const float*  dw = dW + (size_t)b * ((TSTEPS - 1) * NDIM) + step * NDIM;
  float acc = 0.0f;
#pragma unroll
  for (int n = 0; n < NDIM; n++) acc += bf2f(g[n]) * dw[n];
  float dt = ts[step + 1] - ts[step];
  float v = 2.0f * y[idx] - yh[idx] + d0[idx] * dt + acc;
  yh_out[idx]  = v;
  yb_out[idx]  = f2bf(v);
  gdw_out[idx] = acc;
}

// corrector: y1 = y + 0.5*((d0+d1)*dt + gdw0 + g1.dw)
__global__ void post_kernel(const bf16_t* __restrict__ g1,
                            const float* __restrict__ dW, int step,
                            const float* __restrict__ ts,
                            const float* __restrict__ y,
                            const float* __restrict__ d0,
                            const float* __restrict__ d1,
                            const float* __restrict__ gdw0,
                            float* __restrict__ y_out) {
  int idx = blockIdx.x * blockDim.x + threadIdx.x;
  if (idx >= BATCH * HDIM) return;
  int b = idx >> 7, h = idx & 127;
  const bf16_t* g  = g1 + (size_t)b * (HDIM * NDIM) + h * NDIM;
  const float*  dw = dW + (size_t)b * ((TSTEPS - 1) * NDIM) + step * NDIM;
  float acc = 0.0f;
#pragma unroll
  for (int n = 0; n < NDIM; n++) acc += bf2f(g[n]) * dw[n];
  float dt = ts[step + 1] - ts[step];
  y_out[idx] = y[idx] + 0.5f * ((d0[idx] + d1[idx]) * dt + gdw0[idx] + acc);
}

// readout for one time index: out[b][t][d] = roW[d,:] . y[b,:] + rob[d]
__global__ void readout_kernel(const float* __restrict__ y,
                               const float* __restrict__ roW,
                               const float* __restrict__ rob,
                               float* __restrict__ out, int t) {
  int idx = blockIdx.x * blockDim.x + threadIdx.x;   // b*16 + d
  if (idx >= BATCH * DOUT) return;
  int b = idx >> 4, d = idx & 15;
  const float* yr = y + (size_t)b * HDIM;
  const float* wr = roW + d * HDIM;
  float acc = rob[d];
#pragma unroll 8
  for (int h = 0; h < HDIM; h++) acc += yr[h] * wr[h];
  out[(size_t)b * (TSTEPS * DOUT) + t * DOUT + d] = acc;
}

// ---------------------------------------------------------------------------
// Host orchestration
// ---------------------------------------------------------------------------
extern "C" void kernel_launch(void* const* d_in, const int* in_sizes, int n_in,
                              void* d_out, int out_size, void* d_ws, size_t ws_size,
                              hipStream_t stream) {
  const float* ts    = (const float*)d_in[0];
  const float* x0    = (const float*)d_in[1];
  const float* dW    = (const float*)d_in[2];
  const float* iW0   = (const float*)d_in[3];
  const float* ib0   = (const float*)d_in[4];
  const float* iW1   = (const float*)d_in[5];
  const float* ib1   = (const float*)d_in[6];
  const float* iW2   = (const float*)d_in[7];
  const float* ib2   = (const float*)d_in[8];
  const float* vW0   = (const float*)d_in[9];
  const float* vb0   = (const float*)d_in[10];
  const float* vW1   = (const float*)d_in[11];
  const float* vb1   = (const float*)d_in[12];
  const float* vW2   = (const float*)d_in[13];
  const float* vb2   = (const float*)d_in[14];
  const float* vscale= (const float*)d_in[15];
  const float* cW0   = (const float*)d_in[16];
  const float* cb0   = (const float*)d_in[17];
  const float* cW1   = (const float*)d_in[18];
  const float* cb1   = (const float*)d_in[19];
  const float* cW2   = (const float*)d_in[20];
  const float* cb2   = (const float*)d_in[21];
  const float* cscale= (const float*)d_in[22];
  const float* roW   = (const float*)d_in[23];
  const float* rob   = (const float*)d_in[24];
  float* out = (float*)d_out;

  // workspace bump allocator
  char* ws = (char*)d_ws;
  size_t off = 0;
  auto alloc = [&](size_t bytes) -> char* {
    char* p = ws + off;
    off += (bytes + 255) & ~(size_t)255;
    return p;
  };

  // bf16 weights (row-major [N][K], matching the reference W layouts)
  bf16_t* w_iW0  = (bf16_t*)alloc(256 * 32  * 2);
  bf16_t* w_iW1  = (bf16_t*)alloc(256 * 256 * 2);
  bf16_t* w_iW2  = (bf16_t*)alloc(128 * 256 * 2);
  bf16_t* w_vW0y = (bf16_t*)alloc(256 * 128 * 2);
  bf16_t* w_vW1  = (bf16_t*)alloc(256 * 256 * 2);
  bf16_t* w_vW2  = (bf16_t*)alloc(128 * 256 * 2);
  bf16_t* w_cW0y = (bf16_t*)alloc(256 * 128 * 2);
  bf16_t* w_cW1  = (bf16_t*)alloc(256 * 256 * 2);
  bf16_t* w_cW2  = (bf16_t*)alloc(4096 * 256 * 2);
  float*  vcol0  = (float*) alloc(256 * 4);
  float*  ccol0  = (float*) alloc(256 * 4);

  // activations
  bf16_t* x0b  = (bf16_t*)alloc((size_t)BATCH * 32 * 2);
  bf16_t* h1b  = (bf16_t*)alloc((size_t)BATCH * WDIM * 2);
  bf16_t* h2b  = (bf16_t*)alloc((size_t)BATCH * WDIM * 2);
  bf16_t* yb   = (bf16_t*)alloc((size_t)BATCH * HDIM * 2);   // bf16 MLP input
  bf16_t* Gb[2];
  Gb[0] = (bf16_t*)alloc((size_t)BATCH * HDIM * NDIM * 2);   // 32 MB, L2-resident
  Gb[1] = (bf16_t*)alloc((size_t)BATCH * HDIM * NDIM * 2);
  float* Y[2];  Y[0]  = (float*)alloc((size_t)BATCH * HDIM * 4);
                Y[1]  = (float*)alloc((size_t)BATCH * HDIM * 4);
  float* YH[2]; YH[0] = (float*)alloc((size_t)BATCH * HDIM * 4);
                YH[1] = (float*)alloc((size_t)BATCH * HDIM * 4);
  float* D[2];  D[0]  = (float*)alloc((size_t)BATCH * HDIM * 4);
                D[1]  = (float*)alloc((size_t)BATCH * HDIM * 4);
  float* gdw0  = (float*)alloc((size_t)BATCH * HDIM * 4);
  (void)ws_size; (void)in_sizes; (void)n_in; (void)out_size;

  auto cvt = [&](bf16_t* dst, const float* src, int n) {
    cvt_kernel<<<(n + 255) / 256, 256, 0, stream>>>(dst, src, n);
  };
  auto gemm = [&](const bf16_t* A, int lda, const bf16_t* Wp, int ldw,
                  const float* bias, const float* tcol, int tidx,
                  const float* scale, int N, int K, int act,
                  float* oF, bf16_t* oB) {
    if (N >= 1024) {   // dominant cW2 GEMM: big wave tiles (32x64, 8 wmma)
      dim3 grid(N / 128, BATCH / 128);
      gemm_bf16_kernel<128, 4, 2, 2, 4><<<grid, 256, 0, stream>>>(
          A, lda, Wp, ldw, bias, tcol, ts, tidx, scale, BATCH, N, K, act, oF, oB, N);
    } else {           // small-N layers: more blocks for occupancy
      dim3 grid(N / 128, BATCH / 64);
      gemm_bf16_kernel<64, 2, 4, 2, 2><<<grid, 256, 0, stream>>>(
          A, lda, Wp, ldw, bias, tcol, ts, tidx, scale, BATCH, N, K, act, oF, oB, N);
    }
  };

  // ---- weight conversion ----
  cvt(w_iW0, iW0, 256 * 32);
  cvt(w_iW1, iW1, 256 * 256);
  cvt(w_iW2, iW2, 128 * 256);
  cvt(w_vW1, vW1, 256 * 256);
  cvt(w_vW2, vW2, 128 * 256);
  cvt(w_cW1, cW1, 256 * 256);
  cvt(w_cW2, cW2, 4096 * 256);
  split_kernel<<<(256 * 128 + 255) / 256, 256, 0, stream>>>(w_vW0y, vcol0, vW0, 256);
  split_kernel<<<(256 * 128 + 255) / 256, 256, 0, stream>>>(w_cW0y, ccol0, cW0, 256);
  cvt(x0b, x0, BATCH * 32);

  const int BH = BATCH * HDIM;

  // ---- init MLP: y0 = relu(relu(x0@iW0^T)@iW1^T)@iW2^T ----
  gemm(x0b, 32,  w_iW0, 32,  ib0, nullptr, 0, nullptr, 256, 32,  1, nullptr, h1b);
  gemm(h1b, 256, w_iW1, 256, ib1, nullptr, 0, nullptr, 256, 256, 1, nullptr, h2b);
  gemm(h2b, 256, w_iW2, 256, ib2, nullptr, 0, nullptr, 128, 256, 0, Y[0], yb);
  copyf_kernel<<<(BH + 255) / 256, 256, 0, stream>>>(YH[0], Y[0], BH);  // yhat0 = y0

  // ---- d0 = drift(ts[0], y0), g0 = diffusion(ts[0], y0) ----
  gemm(yb,  128, w_vW0y, 128, vb0, vcol0, 0, nullptr, 256, 128, 2, nullptr, h1b);
  gemm(h1b, 256, w_vW1,  256, vb1, nullptr, 0, nullptr, 256, 256, 2, nullptr, h2b);
  gemm(h2b, 256, w_vW2,  256, vb2, nullptr, 0, vscale, 128, 256, 2, D[0], nullptr);
  gemm(yb,  128, w_cW0y, 128, cb0, ccol0, 0, nullptr, 256, 128, 2, nullptr, h1b);
  gemm(h1b, 256, w_cW1,  256, cb1, nullptr, 0, nullptr, 256, 256, 2, nullptr, h2b);
  gemm(h2b, 256, w_cW2,  256, cb2, nullptr, 0, cscale, 4096, 256, 2, nullptr, Gb[0]);

  readout_kernel<<<(BATCH * DOUT + 255) / 256, 256, 0, stream>>>(Y[0], roW, rob, out, 0);

  // ---- 63 Heun steps ----
  int cur = 0;
  for (int i = 0; i < TSTEPS - 1; i++) {
    int nxt = 1 - cur;
    // predictor: yhat1 -> YH[nxt] (f32) + yb (bf16), gdw0
    pre_kernel<<<(BH + 255) / 256, 256, 0, stream>>>(
        Gb[cur], dW, i, ts, Y[cur], YH[cur], D[cur], YH[nxt], yb, gdw0);
    // drift(t1, yhat1) -> D[nxt]
    gemm(yb,  128, w_vW0y, 128, vb0, vcol0, i + 1, nullptr, 256, 128, 2, nullptr, h1b);
    gemm(h1b, 256, w_vW1,  256, vb1, nullptr, 0, nullptr, 256, 256, 2, nullptr, h2b);
    gemm(h2b, 256, w_vW2,  256, vb2, nullptr, 0, vscale, 128, 256, 2, D[nxt], nullptr);
    // diffusion(t1, yhat1) -> Gb[nxt] (bf16)
    gemm(yb,  128, w_cW0y, 128, cb0, ccol0, i + 1, nullptr, 256, 128, 2, nullptr, h1b);
    gemm(h1b, 256, w_cW1,  256, cb1, nullptr, 0, nullptr, 256, 256, 2, nullptr, h2b);
    gemm(h2b, 256, w_cW2,  256, cb2, nullptr, 0, cscale, 4096, 256, 2, nullptr, Gb[nxt]);
    // corrector: y1 -> Y[nxt]
    post_kernel<<<(BH + 255) / 256, 256, 0, stream>>>(
        Gb[nxt], dW, i, ts, Y[cur], D[cur], D[nxt], gdw0, Y[nxt]);
    // readout at time index i+1
    readout_kernel<<<(BATCH * DOUT + 255) / 256, 256, 0, stream>>>(
        Y[nxt], roW, rob, out, i + 1);
    cur = nxt;
  }
}
